// GraphUNet_33397665694061
// MI455X (gfx1250) — matrix operand order repaired and checked by reading, not compiled, not run on hardware
//
#include <hip/hip_runtime.h>
#include <hip/hip_bf16.h>

typedef float    v8f  __attribute__((ext_vector_type(8)));
typedef float    v2f  __attribute__((ext_vector_type(2)));
typedef _Float16 v16h __attribute__((ext_vector_type(16)));
typedef _Float16 v8h  __attribute__((ext_vector_type(8)));

// ---------------------------------------------------------------------------
// f32 WMMA GEMM: C[M,N] = A[M,K] @ B[K,N], epilogue per template:
//   GCN=false: C = acc (+bias) (relu)
//   GCN=true : C = dinv[r] * (acc + B[r][c]) + bias  (relu)   [M==K assumed]
// Wave computes 32x32 (2x2 of 16x16) via V_WMMA_F32_16X16X4_F32.
// Full-tile waves take a guard-free inner loop.
// ---------------------------------------------------------------------------
template<bool GCN>
__global__ __launch_bounds__(128)
void gemm_f32_wmma(const float* __restrict__ A, int lda,
                   const float* __restrict__ B, int ldb,
                   float* __restrict__ C, int ldc,
                   int M, int N, int K,
                   const float* __restrict__ bias, int relu,
                   const float* __restrict__ dinv)
{
    const int wave = blockIdx.x * (blockDim.x >> 5) + (threadIdx.x >> 5);
    const int lane = threadIdx.x & 31;
    const int tmCnt = (M + 31) >> 5, tnCnt = (N + 31) >> 5;
    if (wave >= tmCnt * tnCnt) return;               // wave-uniform
    const int tm = wave / tnCnt, tn = wave % tnCnt;
    const int row0 = tm << 5, col0 = tn << 5;
    const int lh = lane >> 4, ll = lane & 15;
    const bool full = (row0 + 32 <= M) && (col0 + 32 <= N);  // wave-uniform

    v8f acc[2][2] = {};
    const int kmain = K & ~3;

    if (full) {
        const float* arow0 = A + (size_t)(row0 + ll) * lda;
        const float* arow1 = A + (size_t)(row0 + 16 + ll) * lda;
        for (int k0 = 0; k0 < kmain; k0 += 4) {
            const int kk = k0 + 2 * lh;              // f32 frag: K = vgpr + 2*(lane>>4)
            v2f a[2], b[2];
            a[0].x = arow0[kk];     a[0].y = arow0[kk + 1];
            a[1].x = arow1[kk];     a[1].y = arow1[kk + 1];
            const float* brow0 = B + (size_t)kk * ldb;
            const float* brow1 = B + (size_t)(kk + 1) * ldb;
            b[0].x = brow0[col0 + ll];      b[0].y = brow1[col0 + ll];
            b[1].x = brow0[col0 + 16 + ll]; b[1].y = brow1[col0 + 16 + ll];
            __builtin_prefetch(arow0 + kk + 4, 0, 3);          // global_prefetch_b8
            __builtin_prefetch(brow0 + ldb * 4 + col0 + ll, 0, 3);
            #pragma unroll
            for (int i = 0; i < 2; ++i)
                #pragma unroll
                for (int j = 0; j < 2; ++j)
                    acc[i][j] = __builtin_amdgcn_wmma_f32_16x16x4_f32(
                        false, a[i], false, b[j], (short)0, acc[i][j], false, false);
        }
        if (kmain < K) {                              // K%4 tail (e.g. K=3 pos GEMM)
            const int kk = kmain + 2 * lh;
            v2f a[2], b[2];
            #pragma unroll
            for (int i = 0; i < 2; ++i) {
                const float* ar = (i == 0) ? arow0 : arow1;
                a[i].x = (kk     < K) ? ar[kk]     : 0.f;
                a[i].y = (kk + 1 < K) ? ar[kk + 1] : 0.f;
            }
            #pragma unroll
            for (int j = 0; j < 2; ++j) {
                const int c = col0 + j * 16 + ll;
                b[j].x = (kk     < K) ? B[(size_t)kk * ldb + c]       : 0.f;
                b[j].y = (kk + 1 < K) ? B[(size_t)(kk + 1) * ldb + c] : 0.f;
            }
            #pragma unroll
            for (int i = 0; i < 2; ++i)
                #pragma unroll
                for (int j = 0; j < 2; ++j)
                    acc[i][j] = __builtin_amdgcn_wmma_f32_16x16x4_f32(
                        false, a[i], false, b[j], (short)0, acc[i][j], false, false);
        }
    } else {
        for (int k0 = 0; k0 < K; k0 += 4) {
            const int kk = k0 + 2 * lh;
            v2f a[2], b[2];
            #pragma unroll
            for (int i = 0; i < 2; ++i) {
                const int r = row0 + i * 16 + ll;
                a[i].x = (r < M && kk     < K) ? A[(size_t)r * lda + kk]     : 0.f;
                a[i].y = (r < M && kk + 1 < K) ? A[(size_t)r * lda + kk + 1] : 0.f;
            }
            #pragma unroll
            for (int j = 0; j < 2; ++j) {
                const int c = col0 + j * 16 + ll;
                b[j].x = (c < N && kk     < K) ? B[(size_t)kk * ldb + c]       : 0.f;
                b[j].y = (c < N && kk + 1 < K) ? B[(size_t)(kk + 1) * ldb + c] : 0.f;
            }
            #pragma unroll
            for (int i = 0; i < 2; ++i)
                #pragma unroll
                for (int j = 0; j < 2; ++j)
                    acc[i][j] = __builtin_amdgcn_wmma_f32_16x16x4_f32(
                        false, a[i], false, b[j], (short)0, acc[i][j], false, false);
        }
    }

    #pragma unroll
    for (int i = 0; i < 2; ++i) {
        #pragma unroll
        for (int j = 0; j < 2; ++j) {
            const int c = col0 + j * 16 + ll;
            if (c >= N) continue;
            const float badd = bias ? bias[c] : 0.f;
            #pragma unroll
            for (int v = 0; v < 8; ++v) {             // C/D: row = v + 8*(lane>>4)
                const int r = row0 + i * 16 + v + 8 * lh;
                if (r < M) {
                    float val = acc[i][j][v];
                    if (GCN) val = (val + B[(size_t)r * ldb + c]) * dinv[r];
                    val += badd;
                    if (relu) val = fmaxf(val, 0.f);
                    C[(size_t)r * ldc + c] = val;
                }
            }
        }
    }
}

// ---------------------------------------------------------------------------
// Augment GEMM: C = Ah @ Ah (f16 in, f32 out). Adjacency entries are small
// integers -> exact in f16/f32-acc. Ah row-major; AhT = Ah^T so B fragments
// are contiguous (b128 loads). V_WMMA_F32_16X16X32_F16, K=32 per step.
// ---------------------------------------------------------------------------
__global__ __launch_bounds__(128)
void aug_gemm_f16(const _Float16* __restrict__ Ah, const _Float16* __restrict__ AhT,
                  float* __restrict__ C, int n)
{
    const int wave = blockIdx.x * (blockDim.x >> 5) + (threadIdx.x >> 5);
    const int lane = threadIdx.x & 31;
    const int tiles = (n + 31) >> 5;
    if (wave >= tiles * tiles) return;
    const int tm = wave / tiles, tn = wave % tiles;
    const int row0 = tm << 5, col0 = tn << 5;
    const int lh = lane >> 4, ll = lane & 15;
    const bool full = (row0 + 32 <= n) && (col0 + 32 <= n);

    v8f acc[2][2] = {};
    const int kmain = n & ~31;

    if (full) {
        const _Float16* ar0 = Ah + (size_t)(row0 + ll) * n;
        const _Float16* ar1 = Ah + (size_t)(row0 + 16 + ll) * n;
        const _Float16* bc0 = AhT + (size_t)(col0 + ll) * n;
        const _Float16* bc1 = AhT + (size_t)(col0 + 16 + ll) * n;
        for (int k0 = 0; k0 < kmain; k0 += 32) {
            // A 16x32 f16 layout: lane holds K = k0+8*lh+[0..7] and k0+16+8*lh+[0..7]
            v8h a0lo = *(const v8h*)(ar0 + k0 + 8 * lh);
            v8h a0hi = *(const v8h*)(ar0 + k0 + 16 + 8 * lh);
            v8h a1lo = *(const v8h*)(ar1 + k0 + 8 * lh);
            v8h a1hi = *(const v8h*)(ar1 + k0 + 16 + 8 * lh);
            // B 32x16: lane holds K = k0+8*lh+[0..7] and k0+16+8*lh+[0..7] for its column
            v8h b0lo = *(const v8h*)(bc0 + k0 + 8 * lh);
            v8h b0hi = *(const v8h*)(bc0 + k0 + 16 + 8 * lh);
            v8h b1lo = *(const v8h*)(bc1 + k0 + 8 * lh);
            v8h b1hi = *(const v8h*)(bc1 + k0 + 16 + 8 * lh);
            __builtin_prefetch(ar0 + k0 + 32, 0, 3);
            __builtin_prefetch(bc0 + k0 + 32, 0, 3);
            v16h a[2], b[2];
            a[0] = __builtin_shufflevector(a0lo, a0hi, 0,1,2,3,4,5,6,7,8,9,10,11,12,13,14,15);
            a[1] = __builtin_shufflevector(a1lo, a1hi, 0,1,2,3,4,5,6,7,8,9,10,11,12,13,14,15);
            b[0] = __builtin_shufflevector(b0lo, b0hi, 0,1,2,3,4,5,6,7,8,9,10,11,12,13,14,15);
            b[1] = __builtin_shufflevector(b1lo, b1hi, 0,1,2,3,4,5,6,7,8,9,10,11,12,13,14,15);
            #pragma unroll
            for (int i = 0; i < 2; ++i)
                #pragma unroll
                for (int j = 0; j < 2; ++j)
                    acc[i][j] = __builtin_amdgcn_wmma_f32_16x16x32_f16(
                        false, a[i], false, b[j], (short)0, acc[i][j], false, false);
        }
    }
    // guarded tail / edge tiles (wave-uniform selection)
    {
        const int kstart = full ? kmain : 0;
        for (int k0 = kstart; k0 < n; k0 += 32) {
            v16h a[2], b[2];
            #pragma unroll
            for (int i = 0; i < 2; ++i) {
                const int r = row0 + i * 16 + ll;
                #pragma unroll
                for (int e = 0; e < 16; ++e) {
                    const int kk = (e < 8) ? (k0 + 8 * lh + e) : (k0 + 16 + 8 * lh + (e - 8));
                    a[i][e] = (r < n && kk < n) ? Ah[(size_t)r * n + kk] : (_Float16)0;
                }
            }
            #pragma unroll
            for (int j = 0; j < 2; ++j) {
                const int c = col0 + j * 16 + ll;
                #pragma unroll
                for (int e = 0; e < 16; ++e) {
                    const int kk = (e < 8) ? (k0 + 8 * lh + e) : (k0 + 16 + 8 * lh + (e - 8));
                    b[j][e] = (c < n && kk < n) ? AhT[(size_t)c * n + kk] : (_Float16)0;
                }
            }
            #pragma unroll
            for (int i = 0; i < 2; ++i)
                #pragma unroll
                for (int j = 0; j < 2; ++j)
                    acc[i][j] = __builtin_amdgcn_wmma_f32_16x16x32_f16(
                        false, a[i], false, b[j], (short)0, acc[i][j], false, false);
        }
    }

    #pragma unroll
    for (int i = 0; i < 2; ++i) {
        #pragma unroll
        for (int j = 0; j < 2; ++j) {
            const int c = col0 + j * 16 + ll;
            if (c >= n) continue;
            #pragma unroll
            for (int v = 0; v < 8; ++v) {
                const int r = row0 + i * 16 + v + 8 * lh;
                if (r < n) C[(size_t)r * n + c] = acc[i][j][v];
            }
        }
    }
}

// ---------------------------------------------------------------------------
// Elementwise / reduction helpers
// ---------------------------------------------------------------------------
__global__ void build_adj_kernel(const int* __restrict__ ei, int E, int n,
                                 float* __restrict__ adj)
{
    int e = blockIdx.x * blockDim.x + threadIdx.x;
    if (e >= E) return;
    atomicAdd(&adj[(size_t)ei[e] * n + ei[E + e]], 1.0f);  // integer sums: deterministic
}

__global__ void aug_to_f16_kernel(const float* __restrict__ adj,
                                  _Float16* __restrict__ ah,
                                  _Float16* __restrict__ ahT, int n)
{
    long long t = (long long)blockIdx.x * blockDim.x + threadIdx.x;
    if (t >= (long long)n * n) return;
    int r = (int)(t / n), c = (int)(t % n);
    _Float16 v = (r == c) ? (_Float16)1.0f : (_Float16)adj[t];
    ah[t] = v;
    ahT[(size_t)c * n + r] = v;
}

__global__ void rowsum_dinv_kernel(const float* __restrict__ adj, int n,
                                   float* __restrict__ dinv)
{
    __shared__ float red[256];
    const int r = blockIdx.x;
    const float* row = adj + (size_t)r * n;
    float s = 0.f;
    for (int c = threadIdx.x; c < n; c += 256) s += row[c];
    red[threadIdx.x] = s; __syncthreads();
    for (int st = 128; st > 0; st >>= 1) {
        if ((int)threadIdx.x < st) red[threadIdx.x] += red[threadIdx.x + st];
        __syncthreads();
    }
    if (threadIdx.x == 0) dinv[r] = rsqrtf(red[0] + 1.0f);  // A = adj + I
}

__global__ void scale_rows_kernel(float* __restrict__ x, const float* __restrict__ dinv, int n)
{
    int t = blockIdx.x * blockDim.x + threadIdx.x;
    if (t >= n * 128) return;
    x[t] *= dinv[t >> 7];
}

__global__ void norm128_kernel(const float* __restrict__ w, float* __restrict__ out)
{
    __shared__ float red[128];
    float v = w[threadIdx.x];
    red[threadIdx.x] = v * v; __syncthreads();
    for (int st = 64; st > 0; st >>= 1) {
        if ((int)threadIdx.x < st) red[threadIdx.x] += red[threadIdx.x + st];
        __syncthreads();
    }
    if (threadIdx.x == 0) out[0] = sqrtf(red[0]);
}

__global__ void score_kernel(const float* __restrict__ x, const float* __restrict__ w,
                             const float* __restrict__ nrm, int n, float* __restrict__ s)
{
    int i = blockIdx.x * blockDim.x + threadIdx.x;
    if (i >= n) return;
    float acc = 0.f;
    const float* row = x + (size_t)i * 128;
    #pragma unroll 8
    for (int h = 0; h < 128; ++h) acc += row[h] * w[h];
    s[i] = tanhf(acc / nrm[0]);
}

// Stable top-k identical to jax.lax.top_k (desc value, ties by index), with
// score tiles staged through LDS.
__global__ void topk_kernel(const float* __restrict__ s, int n, int k,
                            int* __restrict__ perm, float* __restrict__ scv)
{
    __shared__ float sh[1024];
    const int i = blockIdx.x * blockDim.x + threadIdx.x;
    const float si = (i < n) ? s[i] : 0.f;
    int rank = 0;
    for (int base = 0; base < n; base += 1024) {
        const int cnt = min(1024, n - base);
        __syncthreads();
        for (int t = threadIdx.x; t < cnt; t += blockDim.x) sh[t] = s[base + t];
        __syncthreads();
        if (i < n)
            for (int j = 0; j < cnt; ++j) {
                float sj = sh[j];
                rank += (sj > si) || (sj == si && (base + j) < i);
            }
    }
    if (i < n && rank < k) { perm[rank] = i; scv[rank] = si; }
}

__global__ void gather_x_kernel(const float* __restrict__ xin, const int* __restrict__ perm,
                                const float* __restrict__ scv, float* __restrict__ xout, int k)
{
    int t = blockIdx.x * blockDim.x + threadIdx.x;
    if (t >= k * 128) return;
    int r = t >> 7, h = t & 127;
    xout[t] = xin[(size_t)perm[r] * 128 + h] * scv[r];
}

__global__ void gather_adj_kernel(const float* __restrict__ A2, int n,
                                  const int* __restrict__ perm,
                                  float* __restrict__ out, int k)
{
    long long t = (long long)blockIdx.x * blockDim.x + threadIdx.x;
    if (t >= (long long)k * k) return;
    int r = (int)(t / k), c = (int)(t % k);
    out[t] = (r == c) ? 0.f : A2[(size_t)perm[r] * n + perm[c]];  // diag removed
}

__global__ void scatter_add_kernel(float* __restrict__ dst, const float* __restrict__ src,
                                   const int* __restrict__ perm, int k)
{
    int t = blockIdx.x * blockDim.x + threadIdx.x;
    if (t >= k * 128) return;
    int r = t >> 7, h = t & 127;
    dst[(size_t)perm[r] * 128 + h] += src[t];   // perm rows unique -> exact unpool
}

__global__ void log_softmax_rows(float* __restrict__ out, int cols)
{
    __shared__ float red[256];
    float* p = out + (size_t)blockIdx.x * cols;
    float m = -INFINITY;
    for (int c = threadIdx.x; c < cols; c += 256) m = fmaxf(m, p[c]);
    red[threadIdx.x] = m; __syncthreads();
    for (int st = 128; st > 0; st >>= 1) {
        if ((int)threadIdx.x < st) red[threadIdx.x] = fmaxf(red[threadIdx.x], red[threadIdx.x + st]);
        __syncthreads();
    }
    m = red[0]; __syncthreads();
    float s = 0.f;
    for (int c = threadIdx.x; c < cols; c += 256) s += expf(p[c] - m);
    red[threadIdx.x] = s; __syncthreads();
    for (int st = 128; st > 0; st >>= 1) {
        if ((int)threadIdx.x < st) red[threadIdx.x] += red[threadIdx.x + st];
        __syncthreads();
    }
    const float lse = logf(red[0]) + m;
    for (int c = threadIdx.x; c < cols; c += 256) p[c] -= lse;
}

// ---------------------------------------------------------------------------
// Host orchestration
// ---------------------------------------------------------------------------
static inline char* bump(char*& p, size_t bytes)
{
    char* r = p;
    p += (bytes + 255) & ~(size_t)255;
    return r;
}

static void gemm_plain(const float* A, int lda, const float* B, int ldb,
                       float* C, int ldc, int M, int N, int K,
                       const float* bias, int relu, hipStream_t s)
{
    int waves = ((M + 31) / 32) * ((N + 31) / 32);
    gemm_f32_wmma<false><<<(waves + 3) / 4, 128, 0, s>>>(A, lda, B, ldb, C, ldc,
                                                         M, N, K, bias, relu, nullptr);
}

static void gemm_gcn(const float* A, int lda, const float* B, int ldb,
                     float* C, int ldc, int M, int N, int K,
                     const float* bias, int relu, const float* dinv, hipStream_t s)
{
    int waves = ((M + 31) / 32) * ((N + 31) / 32);
    gemm_f32_wmma<true><<<(waves + 3) / 4, 128, 0, s>>>(A, lda, B, ldb, C, ldc,
                                                        M, N, K, bias, relu, dinv);
}

extern "C" void kernel_launch(void* const* d_in, const int* in_sizes, int n_in,
                              void* d_out, int out_size, void* d_ws, size_t ws_size,
                              hipStream_t stream)
{
    (void)in_sizes; (void)n_in; (void)out_size; (void)ws_size;
    const int N0 = 5000, H = 128, IND = 3, OUTD = 6890, E = 80000;
    const int K1 = 2000, K2 = 1000, K3 = 500;

    const float* pos = (const float*)d_in[0];
    const int*   ei  = (const int*)d_in[1];
    const float *Wd0=(const float*)d_in[2],  *bd0=(const float*)d_in[3];
    const float *Wd1=(const float*)d_in[4],  *bd1=(const float*)d_in[5];
    const float *Wd2=(const float*)d_in[6],  *bd2=(const float*)d_in[7];
    const float *Wd3=(const float*)d_in[8],  *bd3=(const float*)d_in[9];
    const float *pw0=(const float*)d_in[10], *pw1=(const float*)d_in[11], *pw2=(const float*)d_in[12];
    const float *Wu0=(const float*)d_in[13], *bu0=(const float*)d_in[14];
    const float *Wu1=(const float*)d_in[15], *bu1=(const float*)d_in[16];
    const float *Wu2=(const float*)d_in[17], *bu2=(const float*)d_in[18];
    const float *W1 =(const float*)d_in[19], *b1 =(const float*)d_in[20];
    const float *W2 =(const float*)d_in[21], *b2 =(const float*)d_in[22];
    const float *W3 =(const float*)d_in[23], *b3 =(const float*)d_in[24];

    // Workspace carve (~336 MB total)
    char* p = (char*)d_ws;
    float*     adj0  = (float*)    bump(p, (size_t)N0 * N0 * 4);
    float*     scr   = (float*)    bump(p, (size_t)N0 * N0 * 4);   // A2 scratch
    _Float16*  ah    = (_Float16*) bump(p, (size_t)N0 * N0 * 2);   // f16 augmented adj
    _Float16*  ahT   = (_Float16*) bump(p, (size_t)N0 * N0 * 2);   // transpose for B frags
    float*     adj1  = (float*)    bump(p, (size_t)K1 * K1 * 4);
    float*     adj2  = (float*)    bump(p, (size_t)K2 * K2 * 4);
    float*     adj3  = (float*)    bump(p, (size_t)K3 * K3 * 4);
    float*     x0    = (float*)    bump(p, (size_t)N0 * H * 4);
    float*     x1    = (float*)    bump(p, (size_t)K1 * H * 4);
    float*     x2    = (float*)    bump(p, (size_t)K2 * H * 4);
    float*     x3    = (float*)    bump(p, (size_t)K3 * H * 4);
    float*     xg    = (float*)    bump(p, (size_t)N0 * H * 4);    // gather / temp
    float*     xu    = (float*)    bump(p, (size_t)N0 * H * 4);    // unpool sum / temp
    float*     xc    = (float*)    bump(p, (size_t)N0 * H * 4);    // current x / temp
    float*     xw    = (float*)    bump(p, (size_t)N0 * H * 4);    // X@W scratch
    float*     dinv0 = (float*)    bump(p, (size_t)N0 * 4);
    float*     dinv1 = (float*)    bump(p, (size_t)K1 * 4);
    float*     dinv2 = (float*)    bump(p, (size_t)K2 * 4);
    float*     dinv3 = (float*)    bump(p, (size_t)K3 * 4);
    float*     score = (float*)    bump(p, (size_t)N0 * 4);
    float*     scv   = (float*)    bump(p, (size_t)K1 * 4);
    int*       perm0 = (int*)      bump(p, (size_t)K1 * 4);
    int*       perm1 = (int*)      bump(p, (size_t)K2 * 4);
    int*       perm2 = (int*)      bump(p, (size_t)K3 * 4);
    float*     wn    = (float*)    bump(p, 16);

    // --- build dense adjacency (must re-zero each call) ---
    hipMemsetAsync(adj0, 0, (size_t)N0 * N0 * 4, stream);
    build_adj_kernel<<<(E + 255) / 256, 256, 0, stream>>>(ei, E, N0, adj0);

    // --- first GCN: x0 = relu(dinv*(adj0 @ (dinv*pos@Wd0) + dinv*pos@Wd0) + bd0) ---
    rowsum_dinv_kernel<<<N0, 256, 0, stream>>>(adj0, N0, dinv0);
    gemm_plain(pos, IND, Wd0, H, xw, H, N0, H, IND, nullptr, 0, stream);
    scale_rows_kernel<<<(N0 * H + 255) / 256, 256, 0, stream>>>(xw, dinv0, N0);
    gemm_gcn(adj0, N0, xw, H, x0, H, N0, H, N0, bd0, 1, dinv0, stream);

    // --- down path: augment -> pool -> GCN ---
    auto down = [&](int n, int k, const float* adj_n, const float* x_in,
                    const float* pw, const float* Wd, const float* bd,
                    float* adj_k, float* dinv_k, int* perm, float* x_out) {
        aug_to_f16_kernel<<<(int)(((size_t)n * n + 255) / 256), 256, 0, stream>>>(adj_n, ah, ahT, n);
        {
            int t = (n + 31) / 32, waves = t * t;
            aug_gemm_f16<<<(waves + 3) / 4, 128, 0, stream>>>(ah, ahT, scr, n);   // A2
        }
        norm128_kernel<<<1, 128, 0, stream>>>(pw, wn);
        score_kernel<<<(n + 255) / 256, 256, 0, stream>>>(x_in, pw, wn, n, score);
        topk_kernel<<<(n + 255) / 256, 256, 0, stream>>>(score, n, k, perm, scv);
        gather_x_kernel<<<(k * H + 255) / 256, 256, 0, stream>>>(x_in, perm, scv, xg, k);
        gather_adj_kernel<<<(int)(((size_t)k * k + 255) / 256), 256, 0, stream>>>(scr, n, perm, adj_k, k);
        rowsum_dinv_kernel<<<k, 256, 0, stream>>>(adj_k, k, dinv_k);
        gemm_plain(xg, H, Wd, H, xw, H, k, H, H, nullptr, 0, stream);
        scale_rows_kernel<<<(k * H + 255) / 256, 256, 0, stream>>>(xw, dinv_k, k);
        gemm_gcn(adj_k, k, xw, H, x_out, H, k, H, k, bd, 1, dinv_k, stream);
    };
    down(N0, K1, adj0, x0, pw0, Wd1, bd1, adj1, dinv1, perm0, x1);
    down(K1, K2, adj1, x1, pw1, Wd2, bd2, adj2, dinv2, perm1, x2);
    down(K2, K3, adj2, x2, pw2, Wd3, bd3, adj3, dinv3, perm2, x3);

    // --- up path: unpool (scatter) + residual -> GCN ---
    auto up = [&](int nbig, int ksm, const float* res, const float* xsmall,
                  const int* perm, const float* adj_b, const float* dinv_b,
                  const float* Wu, const float* bu, float* xout, int relu) {
        hipMemcpyAsync(xu, res, (size_t)nbig * H * 4, hipMemcpyDeviceToDevice, stream);
        scatter_add_kernel<<<(ksm * H + 255) / 256, 256, 0, stream>>>(xu, xsmall, perm, ksm);
        gemm_plain(xu, H, Wu, H, xw, H, nbig, H, H, nullptr, 0, stream);
        scale_rows_kernel<<<(nbig * H + 255) / 256, 256, 0, stream>>>(xw, dinv_b, nbig);
        gemm_gcn(adj_b, nbig, xw, H, xout, H, nbig, H, nbig, bu, relu, dinv_b, stream);
    };
    up(K2, K3, x2, x3, perm2, adj2, dinv2, Wu0, bu0, xc, 1);   // 1000 nodes
    up(K1, K2, x1, xc, perm1, adj1, dinv1, Wu1, bu1, xg, 1);   // 2000 nodes
    up(N0, K1, x0, xg, perm0, adj0, dinv0, Wu2, bu2, xc, 0);   // 5000 nodes, no relu

    // --- head: (relu(x@W1+b1))@W2+b2, @W3+b3, log_softmax rows ---
    gemm_plain(xc, H, W1, H, xw, H, N0, H, H, b1, 1, stream);
    gemm_plain(xw, H, W2, H, xu, H, N0, H, H, b2, 0, stream);
    gemm_plain(xu, H, W3, OUTD, (float*)d_out, OUTD, N0, OUTD, H, b3, 0, stream);
    log_softmax_rows<<<N0, 256, 0, stream>>>((float*)d_out, OUTD);
}
